// CrossAttentionBlock_32727650796063
// MI455X (gfx1250) — matrix operand order repaired and checked by reading, not compiled, not run on hardware
//
#include <hip/hip_runtime.h>
#include <hip/hip_bf16.h>
#include <math.h>

#define B_   4
#define C_   384
#define N_   4096
#define NH_  8
#define HD_  48
#define P_   256
#define C4_  1536
#define QC_  1152   // 3*C  (q | k | v)

typedef __attribute__((ext_vector_type(16))) __bf16   v16bf;
typedef __attribute__((ext_vector_type(8)))  float    v8f;
typedef __attribute__((ext_vector_type(4)))  float    v4f;
typedef __attribute__((ext_vector_type(4)))  unsigned int v4u;
typedef __attribute__((ext_vector_type(4)))  int      v4i;
typedef __attribute__((ext_vector_type(8)))  int      v8i;

// One WMMA operand fragment: 16 bf16 across 8 VGPRs
union Frag {
    v16bf v;
    __bf16 e[16];
    unsigned short h[16];
    v4u q[2];
};

static __device__ __forceinline__ __bf16 tobf(float f) { return (__bf16)f; }  // hw v_cvt, RNE

// 16-bit A-matrix 16x32 layout (ISA 7.12.2): slot i = 2*vgpr + half
// slots 0..7  -> k = klo8 + i        (klo8 = lane>=16 ? 8 : 0)
// slots 8..15 -> k = 16 + klo8 + i-8
static __device__ __forceinline__ int ka_idx(int i, int lane) {
    return ((i >= 8) ? 16 : 0) + ((lane >= 16) ? 8 : 0) + (((i >> 1) & 3) << 1) + (i & 1);
}
// 16-bit B-matrix 32x16 layout: same k order as A (slots 0..7 -> klo8 + i, slots 8..15 -> 16 + klo8 + i-8)
static __device__ __forceinline__ int kb_idx(int i, int lane) {
    return ((i >= 8) ? 16 : 0) + ((lane >= 16) ? 8 : 0) + (i & 7);
}

static __device__ __forceinline__ v8f wmma_bf16(const Frag& a, const Frag& b, v8f c) {
    return __builtin_amdgcn_wmma_f32_16x16x32_bf16(false, a.v, false, b.v, (short)0, c, false, false);
}
static __device__ __forceinline__ v8f vzero8() {
    v8f z = {0.f,0.f,0.f,0.f,0.f,0.f,0.f,0.f};
    return z;
}

// ---------------- Kernel 1: QKV = xt @ [Wq | Wkv] ----------------
__global__ __launch_bounds__(32) void k_qkv_gemm(const float* __restrict__ x,
                                                 const float* __restrict__ Wq,
                                                 const float* __restrict__ Wkv,
                                                 float* __restrict__ qkv) {
    const int lane = threadIdx.x & 31;
    const int m0 = blockIdx.x * 16;
    const int j0 = blockIdx.y * 16;
    const int mrow = m0 + (lane & 15);
    const int bb = mrow >> 12;
    const int nn = mrow & 4095;
    const int col = j0 + (lane & 15);
    const float* wbase = (col < C_) ? (Wq + col) : (Wkv + (col - C_));
    const int wstride  = (col < C_) ? C_ : (2 * C_);
    v8f acc = vzero8();
    for (int kt = 0; kt < C_ / 32; ++kt) {
        const int kb = kt * 32;
        __builtin_prefetch(wbase + (kb + 32) * wstride, 0, 1);
        Frag af, bf;
        #pragma unroll
        for (int i = 0; i < 16; ++i) {
            int k = kb + ka_idx(i, lane);
            af.e[i] = tobf(x[(bb * C_ + k) * N_ + nn]);   // transposed A gather (coalesces over lanes)
        }
        #pragma unroll
        for (int i = 0; i < 16; ++i) {
            bf.e[i] = tobf(wbase[(kb + kb_idx(i, lane)) * wstride]);
        }
        acc = wmma_bf16(af, bf, acc);
    }
    const int hi8 = (lane >= 16) ? 8 : 0;
    #pragma unroll
    for (int r = 0; r < 8; ++r) qkv[(m0 + r + hi8) * QC_ + col] = acc[r];
}

// ---------------- Kernel 2: L2 norm of q over token axis ----------------
__global__ __launch_bounds__(256) void k_qnorm(const float* __restrict__ qkv,
                                               float* __restrict__ qscale) {
    __shared__ float red[256];
    const int bc = blockIdx.x;
    const int b = bc / C_;
    const int c = bc % C_;
    float ss = 0.f;
    for (int n = threadIdx.x; n < N_; n += 256) {
        float v = qkv[(b * N_ + n) * QC_ + c];
        ss += v * v;
    }
    red[threadIdx.x] = ss;
    __syncthreads();
    for (int s = 128; s > 0; s >>= 1) {
        if (threadIdx.x < s) red[threadIdx.x] += red[threadIdx.x + s];
        __syncthreads();
    }
    if (threadIdx.x == 0) qscale[bc] = 1.f / fmaxf(sqrtf(red[0]), 1e-12f);
}

// ---------------- Kernel 3: kp/vp = K^T @ EF , V^T @ EF (bf16 out) ----------------
__global__ __launch_bounds__(32) void k_kvproj(const float* __restrict__ qkv,
                                               const float* __restrict__ EF,
                                               __bf16* __restrict__ kpvp) {
    const int lane = threadIdx.x & 31;
    const int c0 = blockIdx.x * 16;
    const int p0 = blockIdx.y * 16;
    const int b = blockIdx.z >> 1;
    const int which = blockIdx.z & 1;
    const int cofs = C_ + which * C_;
    const int crow = c0 + (lane & 15);
    const int pcol = p0 + (lane & 15);
    v8f acc = vzero8();
    for (int kt = 0; kt < N_ / 32; ++kt) {
        const int kb = kt * 32;
        __builtin_prefetch(EF + (kb + 32) * P_ + pcol, 0, 1);
        Frag af, bf;
        #pragma unroll
        for (int i = 0; i < 16; ++i) {
            int n = kb + ka_idx(i, lane);
            af.e[i] = tobf(qkv[(b * N_ + n) * QC_ + cofs + crow]);
        }
        #pragma unroll
        for (int i = 0; i < 16; ++i) {
            int n = kb + kb_idx(i, lane);
            bf.e[i] = tobf(EF[n * P_ + pcol]);
        }
        acc = wmma_bf16(af, bf, acc);
    }
    const int hi8 = (lane >= 16) ? 8 : 0;
    #pragma unroll
    for (int r = 0; r < 8; ++r)
        kpvp[((b * 2 + which) * C_ + (c0 + r + hi8)) * P_ + pcol] = tobf(acc[r]);
}

// ---------------- Kernel 4: fused attention per (b,h) ----------------
// LDS layout (dynamic):
//   kp_raw [48][256] bf16 @ 0        (TDM destination)
//   vp_l   [48][256] bf16 @ 24576    (TDM destination; O-GEMM B operand, K=p contiguous)
//   kpT    [256][64] bf16 @ 49152    (transposed kp, zero-padded c=48..63; S-GEMM B operand)
//   S      [8][16][256] f32 @ 81920
//   stats  [8][16][2]  f32 @ 212992
__global__ __launch_bounds__(256) void k_attn(const float* __restrict__ qkv,
                                              const float* __restrict__ qscale,
                                              const __bf16* __restrict__ kpvp,
                                              const float* __restrict__ temperature,
                                              float* __restrict__ ln_in) {
    extern __shared__ char smem[];
    __bf16* kp_raw = (__bf16*)(smem);
    __bf16* vp_l   = (__bf16*)(smem + 24576);
    __bf16* kpT    = (__bf16*)(smem + 49152);
    float*  S      = (float*)(smem + 81920);
    float*  stats  = (float*)(smem + 212992);

    const int bh = blockIdx.x;
    const int b = bh >> 3, h = bh & 7;
    const int tid = threadIdx.x;
    const int w = tid >> 5, lane = tid & 31;
    const float temp = temperature[h];

    const __bf16* kp_src = kpvp + ((size_t)(b * 2 + 0) * C_ + h * HD_) * P_;
    const __bf16* vp_src = kpvp + ((size_t)(b * 2 + 1) * C_ + h * HD_) * P_;

#if __has_builtin(__builtin_amdgcn_tensor_load_to_lds) && __has_builtin(__builtin_amdgcn_s_wait_tensorcnt)
    if (w == 0) {
        // Tensor DMA: contiguous 48*256 bf16 tile (12288 elements, data_size=2B)
        const unsigned ldsbase = __builtin_amdgcn_groupstaticsize();
        const v8i g1 = { 0x00010000,   // data_size=2B (bits17:16=1)
                         0x30000000,   // tensor_dim0[15:0]=12288 << 16
                         0x00010000,   // tensor_dim0 hi=0 | tensor_dim1=1 << 16
                         0x30000000,   // tile_dim0=12288 << 16
                         1,            // tile_dim1=1
                         12288, 0, 0 };// tensor_dim0_stride=12288
        const v4i z4 = {0, 0, 0, 0};
        unsigned long long gk = (unsigned long long)(size_t)kp_src;
        unsigned long long gv = (unsigned long long)(size_t)vp_src;
        v4u g0k = { 1u, ldsbase + 0u,
                    (unsigned)gk, (unsigned)((gk >> 32) & 0x01FFFFFFull) | 0x80000000u };
        v4u g0v = { 1u, ldsbase + 24576u,
                    (unsigned)gv, (unsigned)((gv >> 32) & 0x01FFFFFFull) | 0x80000000u };
#if defined(__clang_major__) && __clang_major__ >= 23
        const v8i z8 = {0,0,0,0,0,0,0,0};
        __builtin_amdgcn_tensor_load_to_lds(g0k, g1, z4, z4, z8, 0);
        __builtin_amdgcn_tensor_load_to_lds(g0v, g1, z4, z4, z8, 0);
#else
        __builtin_amdgcn_tensor_load_to_lds(g0k, g1, z4, z4, 0);
        __builtin_amdgcn_tensor_load_to_lds(g0v, g1, z4, z4, 0);
#endif
        __builtin_amdgcn_s_wait_tensorcnt(0);
    }
#else
    for (int idx = tid; idx < HD_ * P_; idx += 256) {
        kp_raw[idx] = kp_src[idx];
        vp_l[idx]   = vp_src[idx];
    }
#endif
    __syncthreads();
    // transpose kp into kpT [p][c], zero-pad c = 48..63 (K padding for d=48 contraction)
    for (int idx = tid; idx < 256 * 64; idx += 256) {
        int p = idx >> 6, c = idx & 63;
        kpT[idx] = (c < HD_) ? kp_raw[c * P_ + p] : (__bf16)0.0f;
    }
    __syncthreads();

    const int hi8   = (lane >= 16) ? 8 : 0;
    const int klo8  = hi8;                  // A-fragment K offset
    const int klo16 = (lane >= 16) ? 16 : 0;// unused; B fragments also use klo8
    const int m = lane & 15;
    float* Sw  = S + (w * 16) * 256;
    float* stw = stats + w * 32;

    for (int it = 0; it < 4; ++it) {
        const int nt = blockIdx.y * 32 + w * 4 + it;
        const int tok = nt * 16 + m;
        // ---- A fragments of qn * temperature (K = 48, padded to 64) ----
        const float* qrow = qkv + ((size_t)(b * N_ + tok)) * QC_ + h * HD_;
        const float* qsc  = qscale + b * C_ + h * HD_;
        Frag a0, a1;
        {
            v4f x0 = *(const v4f*)(qrow + klo8);
            v4f x1 = *(const v4f*)(qrow + klo8 + 4);
            v4f x2 = *(const v4f*)(qrow + 16 + klo8);
            v4f x3 = *(const v4f*)(qrow + 20 + klo8);
            v4f s0 = *(const v4f*)(qsc + klo8);
            v4f s1 = *(const v4f*)(qsc + klo8 + 4);
            v4f s2 = *(const v4f*)(qsc + 16 + klo8);
            v4f s3 = *(const v4f*)(qsc + 20 + klo8);
            #pragma unroll
            for (int i = 0; i < 4; ++i) {
                a0.e[i]      = tobf(x0[i] * s0[i] * temp);
                a0.e[4 + i]  = tobf(x1[i] * s1[i] * temp);
                a0.e[8 + i]  = tobf(x2[i] * s2[i] * temp);
                a0.e[12 + i] = tobf(x3[i] * s3[i] * temp);
            }
            v4f y0 = *(const v4f*)(qrow + 32 + klo8);
            v4f y1 = *(const v4f*)(qrow + 36 + klo8);
            v4f t0 = *(const v4f*)(qsc + 32 + klo8);
            v4f t1 = *(const v4f*)(qsc + 36 + klo8);
            #pragma unroll
            for (int i = 0; i < 4; ++i) {
                a1.e[i]     = tobf(y0[i] * t0[i] * temp);
                a1.e[4 + i] = tobf(y1[i] * t1[i] * temp);
                a1.e[8 + i]  = (__bf16)0.0f;   // c >= 48: zero padding
                a1.e[12 + i] = (__bf16)0.0f;
            }
        }
        // ---- S = qn @ kp over 16 P-tiles ----
        #pragma unroll 1
        for (int pt = 0; pt < 16; ++pt) {
            const int p = pt * 16 + m;
            Frag b0, b1;
            b0.q[0] = *(const v4u*)(kpT + p * 64 + klo8);       // c = klo8 .. klo8+7
            b0.q[1] = *(const v4u*)(kpT + p * 64 + 16 + klo8);
            b1.q[0] = *(const v4u*)(kpT + p * 64 + 32 + klo8);
            b1.q[1] = *(const v4u*)(kpT + p * 64 + 48 + klo8);
            v8f acc = vzero8();
            acc = wmma_bf16(a0, b0, acc);
            acc = wmma_bf16(a1, b1, acc);
            #pragma unroll
            for (int r = 0; r < 8; ++r) Sw[(r + hi8) * 256 + p] = acc[r];
        }
        asm volatile("s_wait_dscnt 0" ::: "memory");
        // ---- softmax stats: lanes 0..15 own one token row each ----
        if (lane < 16) {
            float mx = -3.4e38f;
            for (int p = 0; p < 256; ++p) mx = fmaxf(mx, Sw[lane * 256 + p]);
            float sum = 0.f;
            for (int p = 0; p < 256; ++p) sum += expf(Sw[lane * 256 + p] - mx);
            stw[lane * 2 + 0] = mx;
            stw[lane * 2 + 1] = 1.f / sum;
        }
        asm volatile("s_wait_dscnt 0" ::: "memory");
        // ---- O = softmax(S) @ vp^T (K = 256) ----
        v8f o0 = vzero8(), o1 = vzero8(), o2 = vzero8();
        const float mx = stw[m * 2 + 0];
        const float dn = stw[m * 2 + 1];
        #pragma unroll 1
        for (int ks = 0; ks < 8; ++ks) {
            const float* srow = Sw + m * 256 + ks * 32;
            v4f p0 = *(const v4f*)(srow + klo8);
            v4f p1 = *(const v4f*)(srow + klo8 + 4);
            v4f p2 = *(const v4f*)(srow + 16 + klo8);
            v4f p3 = *(const v4f*)(srow + 20 + klo8);
            Frag ap;
            #pragma unroll
            for (int i = 0; i < 4; ++i) {
                ap.e[i]      = tobf(expf(p0[i] - mx) * dn);
                ap.e[4 + i]  = tobf(expf(p1[i] - mx) * dn);
                ap.e[8 + i]  = tobf(expf(p2[i] - mx) * dn);
                ap.e[12 + i] = tobf(expf(p3[i] - mx) * dn);
            }
            #pragma unroll
            for (int dt = 0; dt < 3; ++dt) {
                const int d = dt * 16 + m;
                Frag bv;
                bv.q[0] = *(const v4u*)(vp_l + d * P_ + ks * 32 + klo8);
                bv.q[1] = *(const v4u*)(vp_l + d * P_ + ks * 32 + 16 + klo8);
                if (dt == 0)      o0 = wmma_bf16(ap, bv, o0);
                else if (dt == 1) o1 = wmma_bf16(ap, bv, o1);
                else              o2 = wmma_bf16(ap, bv, o2);
            }
        }
        // ---- scrambled scatter (faithful raw reshape: f = d*32768 + h*4096 + tok) ----
        float* lb = ln_in + (size_t)b * (N_ * C_);
        #pragma unroll
        for (int r = 0; r < 8; ++r) {
            const int tokr = nt * 16 + r + hi8;
            lb[(0 * 16 + m) * 32768 + h * 4096 + tokr] = o0[r];
            lb[(1 * 16 + m) * 32768 + h * 4096 + tokr] = o1[r];
            lb[(2 * 16 + m) * 32768 + h * 4096 + tokr] = o2[r];
        }
    }
}

// ---------------- Kernel 5: LayerNorm stats ----------------
__global__ __launch_bounds__(256) void k_lnstats(const float* __restrict__ ln_in,
                                                 float* __restrict__ stats) {
    const int row = blockIdx.x * 8 + (threadIdx.x >> 5);
    const int lane = threadIdx.x & 31;
    float vals[12];
    float s = 0.f;
    #pragma unroll
    for (int i = 0; i < 12; ++i) {
        vals[i] = ln_in[row * C_ + lane + i * 32];
        s += vals[i];
    }
    #pragma unroll
    for (int off = 16; off > 0; off >>= 1) s += __shfl_xor(s, off, 32);
    const float mu = s / (float)C_;
    float vs = 0.f;
    #pragma unroll
    for (int i = 0; i < 12; ++i) { float d = vals[i] - mu; vs += d * d; }
    #pragma unroll
    for (int off = 16; off > 0; off >>= 1) vs += __shfl_xor(vs, off, 32);
    if (lane == 0) {
        stats[row * 2 + 0] = mu;
        stats[row * 2 + 1] = rsqrtf(vs / (float)C_ + 1e-5f);
    }
}

// ---------------- Kernel 6: h1 = GELU(LN(o) @ w1 + b1), bf16 out ----------------
__global__ __launch_bounds__(32) void k_mlp1(const float* __restrict__ ln_in,
                                             const float* __restrict__ stats,
                                             const float* __restrict__ gamma,
                                             const float* __restrict__ beta,
                                             const float* __restrict__ w1,
                                             const float* __restrict__ b1,
                                             __bf16* __restrict__ h1) {
    const int lane = threadIdx.x & 31;
    const int m0 = blockIdx.x * 16;
    const int j0 = blockIdx.y * 16;
    const int mrow = m0 + (lane & 15);
    const int col = j0 + (lane & 15);
    const int klo8  = (lane >= 16) ? 8 : 0;
    const float mu = stats[mrow * 2 + 0];
    const float rs = stats[mrow * 2 + 1];
    v8f acc = vzero8();
    for (int kt = 0; kt < C_ / 32; ++kt) {
        const int kb = kt * 32;
        __builtin_prefetch(w1 + (kb + 32) * C4_ + col, 0, 1);
        const float* arow = ln_in + (size_t)mrow * C_ + kb;
        v4f x0 = *(const v4f*)(arow + klo8);
        v4f x1 = *(const v4f*)(arow + klo8 + 4);
        v4f x2 = *(const v4f*)(arow + 16 + klo8);
        v4f x3 = *(const v4f*)(arow + 20 + klo8);
        v4f g0 = *(const v4f*)(gamma + kb + klo8);
        v4f g1 = *(const v4f*)(gamma + kb + klo8 + 4);
        v4f g2 = *(const v4f*)(gamma + kb + 16 + klo8);
        v4f g3 = *(const v4f*)(gamma + kb + 20 + klo8);
        v4f e0 = *(const v4f*)(beta + kb + klo8);
        v4f e1 = *(const v4f*)(beta + kb + klo8 + 4);
        v4f e2 = *(const v4f*)(beta + kb + 16 + klo8);
        v4f e3 = *(const v4f*)(beta + kb + 20 + klo8);
        Frag af, bf;
        #pragma unroll
        for (int i = 0; i < 4; ++i) {
            af.e[i]      = tobf((x0[i] - mu) * rs * g0[i] + e0[i]);
            af.e[4 + i]  = tobf((x1[i] - mu) * rs * g1[i] + e1[i]);
            af.e[8 + i]  = tobf((x2[i] - mu) * rs * g2[i] + e2[i]);
            af.e[12 + i] = tobf((x3[i] - mu) * rs * g3[i] + e3[i]);
        }
        #pragma unroll
        for (int i = 0; i < 16; ++i) {
            bf.e[i] = tobf(w1[(kb + kb_idx(i, lane)) * C4_ + col]);
        }
        acc = wmma_bf16(af, bf, acc);
    }
    const int hi8 = (lane >= 16) ? 8 : 0;
    const float bias = b1[col];
    #pragma unroll
    for (int r = 0; r < 8; ++r) {
        float v = acc[r] + bias;
        float g = 0.5f * v * (1.f + erff(v * 0.70710678118654752f));   // exact GELU
        h1[(m0 + r + hi8) * C4_ + col] = tobf(g);
    }
}

// ---------------- Kernel 7: out = y + h1 @ w2 + b2, write [B,C,N] ----------------
__global__ __launch_bounds__(32) void k_mlp2(const __bf16* __restrict__ h1,
                                             const float* __restrict__ w2,
                                             const float* __restrict__ b2,
                                             const float* __restrict__ y,
                                             float* __restrict__ out) {
    const int lane = threadIdx.x & 31;
    const int m0 = blockIdx.x * 16;
    const int j0 = blockIdx.y * 16;
    const int mrow = m0 + (lane & 15);
    const int col = j0 + (lane & 15);
    const int klo8 = (lane >= 16) ? 8 : 0;
    v8f acc = vzero8();
    for (int kt = 0; kt < C4_ / 32; ++kt) {
        const int kb = kt * 32;
        __builtin_prefetch(w2 + (kb + 32) * C_ + col, 0, 1);
        const __bf16* arow = h1 + (size_t)mrow * C4_ + kb;
        Frag af, bf;
        af.q[0] = *(const v4u*)(arow + klo8);        // k = klo8 .. klo8+7 (bf16, contiguous)
        af.q[1] = *(const v4u*)(arow + 16 + klo8);   // k = 16+klo8 ..
        #pragma unroll
        for (int i = 0; i < 16; ++i) {
            bf.e[i] = tobf(w2[(kb + kb_idx(i, lane)) * C_ + col]);
        }
        acc = wmma_bf16(af, bf, acc);
    }
    const int hi8 = (lane >= 16) ? 8 : 0;
    const float bias = b2[col];
    #pragma unroll
    for (int r = 0; r < 8; ++r) {
        const int row = m0 + r + hi8;
        const int b = row >> 12;
        const int n = row & 4095;
        const size_t idx = ((size_t)(b * C_ + col)) * N_ + n;   // [B,C,N]
        out[idx] = y[idx] + acc[r] + bias;
    }
}

extern "C" void kernel_launch(void* const* d_in, const int* in_sizes, int n_in,
                              void* d_out, int out_size, void* d_ws, size_t ws_size,
                              hipStream_t stream) {
    const float* x     = (const float*)d_in[0];
    const float* y     = (const float*)d_in[1];
    const float* Wq    = (const float*)d_in[2];
    const float* Wkv   = (const float*)d_in[3];
    const float* EF    = (const float*)d_in[4];
    const float* temp  = (const float*)d_in[5];
    const float* gamma = (const float*)d_in[6];
    const float* beta  = (const float*)d_in[7];
    const float* w1    = (const float*)d_in[8];
    const float* b1    = (const float*)d_in[9];
    const float* w2    = (const float*)d_in[10];
    const float* b2    = (const float*)d_in[11];
    float* outp = (float*)d_out;

    char* ws = (char*)d_ws;
    float*  qkv    = (float*)(ws + 0);                  // 16384*1152*4 = 75,497,472
    float*  qscale = (float*)(ws + 75497472);           // 1536*4
    __bf16* kpvp   = (__bf16*)(ws + 75503616);          // 4*2*384*256*2 = 1,572,864
    float*  ln_in  = (float*)(ws + 77076480);           // 16384*384*4 = 25,165,824
    float*  stats  = (float*)(ws + 102242304);          // 16384*2*4 = 131,072
    __bf16* h1     = (__bf16*)(ws + 102373376);         // 16384*1536*2 = 50,331,648

    const int attn_lds = 214016;
    hipFuncSetAttribute(reinterpret_cast<const void*>(k_attn),
                        hipFuncAttributeMaxDynamicSharedMemorySize, attn_lds);

    k_qkv_gemm<<<dim3(N_ * B_ / 16, QC_ / 16), 32, 0, stream>>>(x, Wq, Wkv, qkv);
    k_qnorm  <<<dim3(B_ * C_), 256, 0, stream>>>(qkv, qscale);
    k_kvproj <<<dim3(C_ / 16, P_ / 16, 2 * B_), 32, 0, stream>>>(qkv, EF, kpvp);
    k_attn   <<<dim3(B_ * NH_, 8), 256, attn_lds, stream>>>(qkv, qscale, kpvp, temp, ln_in);
    k_lnstats<<<dim3(B_ * N_ / 8), 256, 0, stream>>>(ln_in, stats);
    k_mlp1   <<<dim3(B_ * N_ / 16, C4_ / 16), 32, 0, stream>>>(ln_in, stats, gamma, beta, w1, b1, h1);
    k_mlp2   <<<dim3(B_ * N_ / 16, C_ / 16), 32, 0, stream>>>(h1, w2, b2, y, outp);
}
